// LSTMScratch_13838384628252
// MI455X (gfx1250) — matrix operand order repaired and checked by reading, not compiled, not run on hardware
//
#include <hip/hip_runtime.h>

// ---------------------------------------------------------------------------
// LSTM (T=512, B=64, D=H=1024) for MI455X / gfx1250.
//   Phase 0: Wx,Wh -> bf16 transposed [4096][1024]; bias concat;
//            inputs -> bf16; H0 -> bf16.
//   Phase 1: Xpre = X @ Wx + b  (one big parallel bf16-WMMA GEMM, f32 accum)
//   Phase 2: 512 sequential step kernels: gates = Xpre[t] + H@Wh, in-register
//            sigmoid/tanh cell update on the WMMA accumulators; H flows in
//            bf16 ping-pong ws buffers.
//   Phase 3: append (H_f, C_f) to d_out.
// GEMM core: WG = 64(M) x 16(hcols) x 4(gates); 8 wave32s = 4 row-tiles x
// 2 K-halves (split-K across waves + LDS reduction) -> 64 WGs per step for
// the sequential recurrence.  Data movement: Tensor Data Mover — the whole
// 64x1024 A operand (128 KB) is one TDM descriptor into LDS; B tiles
// (4 gate planes x 16 rows x 128 B, one 3D descriptor per K-step) are
// double-buffered and software-pipelined with s_wait_tensorcnt 1.
// Manual-copy fallback when the TDM builtin is unavailable.
// ---------------------------------------------------------------------------

#define TSTEPS 512
#define BATCH  64
#define HID    1024
#define NGATE  4096   // 4*HID

typedef __attribute__((ext_vector_type(16))) __bf16 v16bf;
typedef __attribute__((ext_vector_type(8)))  float  v8f;
typedef unsigned int v4u __attribute__((ext_vector_type(4)));
typedef int          v4i __attribute__((ext_vector_type(4)));
typedef int          v8i __attribute__((ext_vector_type(8)));

#if defined(__has_builtin)
#  if __has_builtin(__builtin_amdgcn_tensor_load_to_lds) && \
      __has_builtin(__builtin_amdgcn_s_wait_tensorcnt)
#    define HAVE_TDM 1
#  endif
#endif
#ifndef HAVE_TDM
#  define HAVE_TDM 0
#endif
#if __has_include(<hip/amd_detail/amd_gfx1250_TDM.h>)
#  define TDM_6ARG 1     // therock-10.0 headers -> 6-arg builtin
#else
#  define TDM_6ARG 0     // ROCm 7.2 -> 5-arg builtin
#endif

union FragBF {
  uint4 q[2];
  v16bf v;
};

__device__ __forceinline__ unsigned short f2bf(float x) {
  unsigned int u = __float_as_uint(x);
  unsigned int r = ((u >> 16) & 1u) + 0x7FFFu;   // round-to-nearest-even
  return (unsigned short)((u + r) >> 16);
}

__device__ __forceinline__ float sigmoidf_(float x) {
  return 1.0f / (1.0f + __expf(-x));
}

#if HAVE_TDM
// TDM tile load, bf16 elements, packed contiguously into LDS at lds_off.
//   cols/rows/planes = tile_dim0/1/2 (planes==0 -> plain 2D tile)
//   row_stride   = tensor_dim0_stride (elements)
//   plane_stride = tensor_dim1_stride (elements)
// Descriptor layout per CDNA5 ISA ch.8 (D# groups 0..3).
__device__ __forceinline__ void tdm_load_tiles(unsigned lds_off,
                                               const void* gsrc,
                                               unsigned cols, unsigned rows,
                                               unsigned planes,
                                               unsigned row_stride,
                                               unsigned long long plane_stride) {
  unsigned long long ga = (unsigned long long)(uintptr_t)gsrc;
  v4u g0 = { 1u,                                   // count=1 (valid), user mode
             lds_off,                              // lds_addr (bytes)
             (unsigned)ga,                         // global_addr[31:0]
             (unsigned)((ga >> 32) & 0x1FFFFFFu)   // global_addr[56:32]
               | 0x80000000u };                    // type=2 ("image")
  v8i g1 = { (int)0x00010000u,                     // data_size=1 -> 2 bytes
             (int)((cols & 0xFFFFu) << 16),        // tensor_dim0 lo16
             (int)((cols >> 16) | ((rows & 0xFFFFu) << 16)),  // dim0 hi | dim1 lo
             (int)((rows >> 16) | ((cols & 0xFFFFu) << 16)),  // dim1 hi | tile_dim0
             (int)((rows & 0xFFFFu) | (planes << 16)),        // tile_dim1 | tile_dim2
             (int)row_stride,                                 // dim0_stride lo32
             (int)(((unsigned)(plane_stride & 0xFFFFull)) << 16), // |dim1_stride lo16
             (int)(unsigned)(plane_stride >> 16) };           // dim1_stride [47:16]
  v4i g2 = { (int)(planes ? planes : 1u), 0, 0, 0 };          // tensor_dim2
  v4i g3 = { 0, 0, 0, 0 };
#if TDM_6ARG
  v8i z8 = {0, 0, 0, 0, 0, 0, 0, 0};
  __builtin_amdgcn_tensor_load_to_lds(g0, g1, g2, g3, z8, 0);
#else
  __builtin_amdgcn_tensor_load_to_lds(g0, g1, g2, g3, 0);
#endif
}
#endif

// ---------------------------------------------------------------------------
// GEMM core: one workgroup (8 wave32s) computes a 64(M) x 16(hcols) x
// 4(gates) tile over K=1024.  Wave (rowT, kh): rows [rowT*16,+16),
// K chunks [i*64+kh*32,+32), i<16.  After the K loop, kh=1 partials are reduced into
// kh=0 waves through LDS; acc[] is then valid for waves wid<4 only.
// Abf: bf16 [64][1024] (whole slab cached in LDS).
// Bt: bf16 [4096][1024] (row = gate*1024 + h, contiguous K).
// Fragment layouts per ISA 7.12.2 (wave32).
// ---------------------------------------------------------------------------
__device__ __forceinline__ void gemm_core_64x16(
    const unsigned short* __restrict__ Abf,
    const unsigned short* __restrict__ Bt,
    int n0,
    unsigned short* As,   // LDS [64*1024] bf16  (128 KB A slab)
    unsigned short* Bs,   // LDS [2][64*64] bf16 (double-buffered B, 16 KB)
    v8f acc[4])
{
  const int tid  = threadIdx.x;
  const int lane = tid & 31;
  const int wid  = tid >> 5;
  const int rowT = wid & 3;        // 16-row tile index
  const int kh   = wid >> 2;       // K half: 0 or 1
  const int half = lane >> 4;
  const int nn   = lane & 15;

#pragma unroll
  for (int g = 0; g < 4; ++g) acc[g] = v8f{0.f,0.f,0.f,0.f,0.f,0.f,0.f,0.f};

  const unsigned short* Bbase = Bt + (size_t)n0 * 1024;

#if HAVE_TDM
  if (wid == 0) {
    // whole A slab: 64 rows x 1024 elements, contiguous (one descriptor)
    tdm_load_tiles((unsigned)(uintptr_t)As, Abf, 1024u, 64u, 0u, 1024u, 0ull);
    // B k-step 0: 3D tile, 4 gate planes x 16 rows x 64 elements
    tdm_load_tiles((unsigned)(uintptr_t)Bs, Bbase,
                   64u, 16u, 4u, 1024u, 1024ull * 1024ull);
  }
  for (int i = 0; i < 16; ++i) {
    if (wid == 0) {
      if (i < 15) {
        tdm_load_tiles((unsigned)(uintptr_t)Bs + (unsigned)(((i + 1) & 1) * 8192),
                       Bbase + (i + 1) * 64,
                       64u, 16u, 4u, 1024u, 1024ull * 1024ull);
        __builtin_amdgcn_s_wait_tensorcnt(1);   // current tile complete
      } else {
        __builtin_amdgcn_s_wait_tensorcnt(0);   // drain
      }
    }
    __syncthreads();

    const int ks = i * 64 + kh * 32;             // this wave's K base
    const unsigned short* Bcur = Bs + (i & 1) * 4096;

    FragBF a;   // A fragment from slab: row = rowT*16 + nn, K-half by half
    {
      const unsigned short* ap = As + (size_t)(rowT * 16 + nn) * 1024 + ks;
      a.q[0] = *(const uint4*)(ap + half * 8);
      a.q[1] = *(const uint4*)(ap + 16 + half * 8);
    }
#pragma unroll
    for (int g = 0; g < 4; ++g) {
      FragBF b;
      const unsigned short* bp = Bcur + (g * 16 + nn) * 64 + kh * 32 + half * 8;
      b.q[0] = *(const uint4*)(bp);
      b.q[1] = *(const uint4*)(bp + 16);
      acc[g] = __builtin_amdgcn_wmma_f32_16x16x32_bf16(
          false, a.v, false, b.v, (short)0, acc[g], false, false);
    }
    __syncthreads();
  }
#else
  // ---- fallback: manual copies ----
#pragma unroll
  for (int j = 0; j < 32; ++j) {          // A slab: 8192 uint4, 32 per thread
    int idx = tid + 256 * j;
    ((uint4*)As)[idx] = ((const uint4*)Abf)[idx];
  }
  for (int i = 0; i < 16; ++i) {
    const int k0 = i * 64;
#pragma unroll
    for (int j = 0; j < 2; ++j) {         // B tile: 512 uint4 slots
      int f = tid + 256 * j;
      int R = f >> 3, q = f & 7;          // 64 rows x 8 uint4
      int gg = R >> 4, nc = R & 15;
      ((uint4*)(Bs + R * 64))[q] =
          ((const uint4*)(Bbase + ((size_t)gg * 1024 + nc) * 1024 + k0))[q];
    }
    if (k0 + 64 < 1024) {
      int R = tid >> 2;
      __builtin_prefetch(Bbase + ((size_t)(R >> 4) * 1024 + (R & 15)) * 1024
                             + k0 + 64, 0, 0);
    }
    __syncthreads();
    const int ks = k0 + kh * 32;
    FragBF a;
    {
      const unsigned short* ap = As + (size_t)(rowT * 16 + nn) * 1024 + ks;
      a.q[0] = *(const uint4*)(ap + half * 8);
      a.q[1] = *(const uint4*)(ap + 16 + half * 8);
    }
#pragma unroll
    for (int g = 0; g < 4; ++g) {
      FragBF b;
      const unsigned short* bp = Bs + (g * 16 + nn) * 64 + kh * 32 + half * 8;
      b.q[0] = *(const uint4*)(bp);
      b.q[1] = *(const uint4*)(bp + 16);
      acc[g] = __builtin_amdgcn_wmma_f32_16x16x32_bf16(
          false, a.v, false, b.v, (short)0, acc[g], false, false);
    }
    __syncthreads();
  }
#endif

  // ---- cross-K reduction: kh=1 partials into kh=0 waves (LDS, 16 KB) ----
  float* red = (float*)Bs;      // B buffers are dead now; lane-contiguous
  if (kh == 1) {
#pragma unroll
    for (int g = 0; g < 4; ++g)
#pragma unroll
      for (int r = 0; r < 8; ++r)
        red[(rowT * 4 + g) * 256 + r * 32 + lane] = acc[g][r];
  }
  __syncthreads();
  if (kh == 0) {
#pragma unroll
    for (int g = 0; g < 4; ++g)
#pragma unroll
      for (int r = 0; r < 8; ++r)
        acc[g][r] += red[(rowT * 4 + g) * 256 + r * 32 + lane];
  }
}

// ---------------------------------------------------------------------------
// Phase 0a: tiled transpose + f32->bf16:  dst[(g*1024+n)*1024+k] = src_g[k][n]
// ---------------------------------------------------------------------------
__global__ void __launch_bounds__(256)
lstm_wtrans_kernel(const float* __restrict__ s0, const float* __restrict__ s1,
                   const float* __restrict__ s2, const float* __restrict__ s3,
                   unsigned short* __restrict__ dst)
{
  __shared__ float tile[32][33];
  const int g = blockIdx.z;
  const float* src = (g == 0) ? s0 : (g == 1) ? s1 : (g == 2) ? s2 : s3;
  const int bx = blockIdx.x;   // n tile
  const int by = blockIdx.y;   // k tile
  const int tid = threadIdx.x;
  {
    int nl = tid & 31, kl = tid >> 5;
#pragma unroll
    for (int j = 0; j < 4; ++j)
      tile[kl + j * 8][nl] = src[(size_t)(by * 32 + kl + j * 8) * 1024 + bx * 32 + nl];
  }
  __syncthreads();
  {
    int kl = tid & 31, nl = tid >> 5;
#pragma unroll
    for (int j = 0; j < 4; ++j) {
      int n = bx * 32 + nl + j * 8;
      dst[((size_t)g * 1024 + n) * 1024 + by * 32 + kl] = f2bf(tile[kl][nl + j * 8]);
    }
  }
}

__global__ void __launch_bounds__(256)
lstm_bias_kernel(const float* __restrict__ bi, const float* __restrict__ bf,
                 const float* __restrict__ bo, const float* __restrict__ bc,
                 float* __restrict__ bias)
{
  int j = blockIdx.x * 256 + threadIdx.x;
  if (j >= NGATE) return;
  int g = j >> 10;
  const float* s = (g == 0) ? bi : (g == 1) ? bf : (g == 2) ? bo : bc;
  bias[j] = s[j & 1023];
}

// Phase 0b: bulk f32 -> bf16 (float4 granular).
__global__ void __launch_bounds__(256)
lstm_cvt_bf16_kernel(const float* __restrict__ src,
                     unsigned short* __restrict__ dst, int n4)
{
  int i = blockIdx.x * 256 + threadIdx.x;
  if (i >= n4) return;
  float4 v = ((const float4*)src)[i];
  uint2 p;
  p.x = (unsigned)f2bf(v.x) | ((unsigned)f2bf(v.y) << 16);
  p.y = (unsigned)f2bf(v.z) | ((unsigned)f2bf(v.w) << 16);
  ((uint2*)dst)[i] = p;
}

// ---------------------------------------------------------------------------
// Phase 1: Xpre[m, g*1024+h] = (X @ Wx)[m, ...] + b    grid = (64, T*B/64)
// ---------------------------------------------------------------------------
__global__ void __launch_bounds__(256)
lstm_pre_gemm_kernel(const unsigned short* __restrict__ Xbf,
                     const unsigned short* __restrict__ WxT,
                     const float* __restrict__ bias,
                     float* __restrict__ Xpre)
{
  __shared__ unsigned short As[64 * 1024];
  __shared__ unsigned short Bs[2 * 64 * 64];
  v8f acc[4];
  const unsigned short* A = Xbf + (size_t)blockIdx.y * 64 * 1024;
  const int n0 = blockIdx.x * 16;
  gemm_core_64x16(A, WxT, n0, As, Bs, acc);

  const int lane = threadIdx.x & 31, wid = threadIdx.x >> 5;
  if (wid >= 4) return;                          // kh=0 waves hold the sums
  const int rowT = wid, half = lane >> 4, nn = lane & 15;
  const int colh = n0 + nn;
#pragma unroll
  for (int g = 0; g < 4; ++g) {
    float bv = bias[g * 1024 + colh];
#pragma unroll
    for (int r = 0; r < 8; ++r) {
      size_t m = (size_t)blockIdx.y * 64 + rowT * 16 + half * 8 + r;
      Xpre[m * NGATE + g * 1024 + colh] = acc[g][r] + bv;
    }
  }
}

// ---------------------------------------------------------------------------
// Phase 2: one recurrence step; H flows through bf16 ping-pong buffers.
// grid = (64)
// ---------------------------------------------------------------------------
__global__ void __launch_bounds__(256)
lstm_step_kernel(const unsigned short* __restrict__ HbfIn,
                 const unsigned short* __restrict__ WhT,
                 const float* __restrict__ XpreT,
                 const float* __restrict__ Cin,
                 float* __restrict__ Cout,
                 float* __restrict__ Hout,
                 unsigned short* __restrict__ HbfOut)
{
  __shared__ unsigned short As[64 * 1024];
  __shared__ unsigned short Bs[2 * 64 * 64];
  v8f acc[4];
  const int n0 = blockIdx.x * 16;
  gemm_core_64x16(HbfIn, WhT, n0, As, Bs, acc);

  const int lane = threadIdx.x & 31, wid = threadIdx.x >> 5;
  if (wid >= 4) return;                          // kh=0 waves hold the sums
  const int rowT = wid, half = lane >> 4, nn = lane & 15;
  const int h = n0 + nn;
#pragma unroll
  for (int r = 0; r < 8; ++r) {
    const int m = rowT * 16 + half * 8 + r;      // batch row
    const float* xp = XpreT + (size_t)m * NGATE + h;
    float I = sigmoidf_(acc[0][r] + xp[0]);
    float F = sigmoidf_(acc[1][r] + xp[1024]);
    float O = sigmoidf_(acc[2][r] + xp[2048]);
    float G = tanhf(acc[3][r] + xp[3072]);
    float c = F * Cin[(size_t)m * HID + h] + I * G;
    float hv = O * tanhf(c);
    Cout[(size_t)m * HID + h] = c;
    Hout[(size_t)m * HID + h] = hv;
    HbfOut[(size_t)m * HID + h] = f2bf(hv);
  }
}

// Phase 3: append H_f (= outputs[T-1]) and C_f.
__global__ void __launch_bounds__(256)
lstm_tail_kernel(const float* __restrict__ Hlast,
                 const float* __restrict__ Cst,
                 float* __restrict__ dst)
{
  int i = blockIdx.x * 256 + threadIdx.x;
  if (i >= BATCH * HID) return;
  dst[i] = Hlast[i];
  dst[BATCH * HID + i] = Cst[i];
}

// ---------------------------------------------------------------------------
extern "C" void kernel_launch(void* const* d_in, const int* in_sizes, int n_in,
                              void* d_out, int out_size, void* d_ws, size_t ws_size,
                              hipStream_t stream)
{
  (void)in_sizes; (void)n_in; (void)out_size; (void)ws_size;

  const float* inputs = (const float*)d_in[0];   // [T,B,D]
  const float* H0     = (const float*)d_in[1];
  const float* C0     = (const float*)d_in[2];
  const float* Wxi = (const float*)d_in[3];
  const float* Wxf = (const float*)d_in[4];
  const float* Wxo = (const float*)d_in[5];
  const float* Wxc = (const float*)d_in[6];
  const float* Whi = (const float*)d_in[7];
  const float* Whf = (const float*)d_in[8];
  const float* Who = (const float*)d_in[9];
  const float* Whc = (const float*)d_in[10];
  const float* bi  = (const float*)d_in[11];
  const float* bf  = (const float*)d_in[12];
  const float* bo  = (const float*)d_in[13];
  const float* bc  = (const float*)d_in[14];

  float* out = (float*)d_out;
  char*  ws  = (char*)d_ws;

  // workspace layout
  const size_t OFF_WXT  = 0;                                // 8 MB bf16
  const size_t OFF_WHT  = (size_t)8 << 20;                  // 8 MB bf16
  const size_t OFF_BIAS = (size_t)16 << 20;                 // 16 KB f32
  const size_t OFF_C    = ((size_t)16 << 20) + (64 << 10);  // 256 KB f32
  const size_t OFF_HBF  = ((size_t)16 << 20) + (384 << 10); // 2 x 128 KB bf16
  const size_t OFF_XBF  = (size_t)17 << 20;                 // 64 MB bf16
  const size_t OFF_XPRE = (size_t)81 << 20;                 // 512 MB f32

  unsigned short* WxT  = (unsigned short*)(ws + OFF_WXT);
  unsigned short* WhT  = (unsigned short*)(ws + OFF_WHT);
  float*          bias = (float*)(ws + OFF_BIAS);
  float*          wsC  = (float*)(ws + OFF_C);
  unsigned short* Hbf0 = (unsigned short*)(ws + OFF_HBF);
  unsigned short* Hbf1 = (unsigned short*)(ws + OFF_HBF + (128 << 10));
  unsigned short* Xbf  = (unsigned short*)(ws + OFF_XBF);
  float*          Xpre = (float*)(ws + OFF_XPRE);

  // Phase 0
  lstm_wtrans_kernel<<<dim3(32, 32, 4), 256, 0, stream>>>(Wxi, Wxf, Wxo, Wxc, WxT);
  lstm_wtrans_kernel<<<dim3(32, 32, 4), 256, 0, stream>>>(Whi, Whf, Who, Whc, WhT);
  lstm_bias_kernel<<<16, 256, 0, stream>>>(bi, bf, bo, bc, bias);
  lstm_cvt_bf16_kernel<<<32768, 256, 0, stream>>>(inputs, Xbf,
                                                  (TSTEPS * BATCH * HID) / 4);
  lstm_cvt_bf16_kernel<<<64, 256, 0, stream>>>(H0, Hbf0, (BATCH * HID) / 4);

  // Phase 1
  lstm_pre_gemm_kernel<<<dim3(HID / 16, (TSTEPS * BATCH) / 64), 256, 0, stream>>>(
      Xbf, WxT, bias, Xpre);

  // Phase 2: sequential recurrence (bf16 H ping-pong; f32 H into d_out).
  const size_t BH = (size_t)BATCH * HID;
  for (int t = 0; t < TSTEPS; ++t) {
    const unsigned short* Hin  = (t & 1) ? Hbf1 : Hbf0;
    unsigned short*       Hnxt = (t & 1) ? Hbf0 : Hbf1;
    const float* Cin = (t == 0) ? C0 : wsC;
    lstm_step_kernel<<<dim3(HID / 16), 256, 0, stream>>>(
        Hin, WhT, Xpre + (size_t)t * BATCH * NGATE, Cin, wsC,
        out + (size_t)t * BH, Hnxt);
  }

  // Phase 3
  lstm_tail_kernel<<<dim3((BATCH * HID) / 256), 256, 0, stream>>>(
      out + (size_t)(TSTEPS - 1) * BH, wsC, out + (size_t)TSTEPS * BH);
}
